// SelfAttention_46299747451321
// MI455X (gfx1250) — matrix operand order repaired and checked by reading, not compiled, not run on hardware
//
#include <hip/hip_runtime.h>
#include <hip/hip_bf16.h>

// Fused causal self-attention for B=2, T=2048, C=2048, H=16, D=128.
// Phase 0: fp32 -> fp16 convert of x and w_qkv.
// Phase 1: QKV GEMM (WMMA f16, f32 accum) with double-buffered
//          global_load_async_to_lds_b128 staging (ASYNCcnt), bias add,
//          writes Q,K row-major f16 and V transposed [B*H, D, T] f16.
// Phase 2: per-wave flash attention (16 q-rows per wave) with WMMA for both
//          Q*K^T and P*V, online softmax via wave32 shuffles.

typedef _Float16 h16;
typedef __attribute__((ext_vector_type(16))) _Float16 v16h;
typedef __attribute__((ext_vector_type(8)))  float    v8f;

#define BDIM 2
#define TDIM 2048
#define CDIM 2048
#define HDIM 16
#define DDIM 128

__device__ __forceinline__ float fast_exp2(float x) {
  return __builtin_amdgcn_exp2f(x);  // v_exp_f32
}

__device__ __forceinline__ v16h frag_ld(const h16* lo, const h16* hi) {
  union { v16h v; uint4 q[2]; } u;
  u.q[0] = *(const uint4*)lo;
  u.q[1] = *(const uint4*)hi;
  return u.v;
}

// Async 16B global -> LDS copy (per-lane addresses), tracked by ASYNCcnt.
__device__ __forceinline__ void async_ld16(uint32_t lds_off, const h16* gptr) {
  asm volatile("global_load_async_to_lds_b128 %0, %1, off"
               :: "v"(lds_off), "v"(gptr)
               : "memory");
}

// ---------------- phase 0: fp32 -> fp16 ----------------
__global__ __launch_bounds__(256) void cvt_f32_to_f16_k(const float* __restrict__ s,
                                                        h16* __restrict__ d, int n4) {
  int i = blockIdx.x * blockDim.x + threadIdx.x;
  if (i < n4) {
    float4 f = *(const float4*)(s + (size_t)i * 4);
    union { h16 h[4]; uint2 u; } o;
    o.h[0] = (h16)f.x; o.h[1] = (h16)f.y; o.h[2] = (h16)f.z; o.h[3] = (h16)f.w;
    *(uint2*)(d + (size_t)i * 4) = o.u;
  }
}

// ---------------- phase 1: QKV projection GEMM ----------------
// out[m,n] = sum_k X[m,k] * W[n,k] + bias[n];  M=4096, N=6144, K=2048.
// Block tile 128x128, K-step 32, double-buffered async LDS staging.
// 8 waves: wave grid 2(M) x 4(N), each wave 64x32.
#define LROW 40                    // padded LDS row stride in halves (80 B)
#define LBUF (128 * LROW)          // halves per buffer

__global__ __launch_bounds__(256) void qkv_gemm_k(const h16* __restrict__ X,
                                                  const h16* __restrict__ W,
                                                  const float* __restrict__ bias,
                                                  h16* __restrict__ Qb,
                                                  h16* __restrict__ Kb,
                                                  h16* __restrict__ Vtb) {
  __shared__ __align__(16) h16 lA[2][LBUF];
  __shared__ __align__(16) h16 lB[2][LBUF];

  const int tid  = threadIdx.x;
  const int lane = tid & 31;
  const int wave = tid >> 5;
  const int half = lane >> 4;
  const int nl   = lane & 15;
  const int mB   = blockIdx.y * 128;
  const int nB   = blockIdx.x * 128;
  const int wm   = wave >> 2;   // 0..1 (64 rows each)
  const int wn   = wave & 3;    // 0..3 (32 cols each)

  // LDS byte offsets of the two staging buffers (generic addr low 32 bits).
  const uint32_t baseA = (uint32_t)(size_t)(&lA[0][0]);
  const uint32_t baseB = (uint32_t)(size_t)(&lB[0][0]);

  // Per-thread staging slots: unit u = tid*2+i -> row u>>2, 8-half col (u&3)*8.
  const int r0 = (tid * 2) >> 2;
  const int c0 = ((tid * 2) & 3) * 8;
  const int r1 = (tid * 2 + 1) >> 2;
  const int c1 = ((tid * 2 + 1) & 3) * 8;

#define ISSUE_TILE(k0, buf)                                                     \
  do {                                                                          \
    uint32_t bo = (uint32_t)(buf) * (LBUF * 2);                                 \
    async_ld16(baseA + bo + (uint32_t)(r0 * LROW + c0) * 2,                     \
               &X[(size_t)(mB + r0) * CDIM + (k0) + c0]);                       \
    async_ld16(baseA + bo + (uint32_t)(r1 * LROW + c1) * 2,                     \
               &X[(size_t)(mB + r1) * CDIM + (k0) + c1]);                       \
    async_ld16(baseB + bo + (uint32_t)(r0 * LROW + c0) * 2,                     \
               &W[(size_t)(nB + r0) * CDIM + (k0) + c0]);                       \
    async_ld16(baseB + bo + (uint32_t)(r1 * LROW + c1) * 2,                     \
               &W[(size_t)(nB + r1) * CDIM + (k0) + c1]);                       \
  } while (0)

  v8f acc[4][2];
#pragma unroll
  for (int i = 0; i < 4; ++i)
#pragma unroll
    for (int j = 0; j < 2; ++j)
      acc[i][j] = (v8f){0.f, 0.f, 0.f, 0.f, 0.f, 0.f, 0.f, 0.f};

  // Prologue: fetch tile 0 into buffer 0.
  ISSUE_TILE(0, 0);

  const int NT = CDIM / 32;  // 64 K-tiles
  for (int it = 0; it < NT; ++it) {
    const int cur = it & 1;

    if (it + 1 < NT) {
      ISSUE_TILE((it + 1) * 32, 1 - cur);
      // 4 newer loads in flight; the 4 older (tile `it`) must be complete.
      asm volatile("s_wait_asynccnt 0x4" ::: "memory");
    } else {
      asm volatile("s_wait_asynccnt 0x0" ::: "memory");
    }
    __syncthreads();  // tile `it` visible to all waves

    const h16* cA = &lA[cur][0];
    const h16* cB = &lB[cur][0];

    v16h bf[2];
#pragma unroll
    for (int fn = 0; fn < 2; ++fn) {
      const h16* p = &cB[(wn * 32 + fn * 16 + nl) * LROW + half * 8];
      bf[fn] = frag_ld(p, p + 16);
    }
#pragma unroll
    for (int fm = 0; fm < 4; ++fm) {
      const h16* p = &cA[(wm * 64 + fm * 16 + nl) * LROW + half * 8];
      v16h af = frag_ld(p, p + 16);
#pragma unroll
      for (int fn = 0; fn < 2; ++fn)
        acc[fm][fn] = __builtin_amdgcn_wmma_f32_16x16x32_f16(
            false, af, false, bf[fn], (short)0, acc[fm][fn], false, false);
    }
    __syncthreads();  // done reading buf[cur] before it is refilled next iter
  }
#undef ISSUE_TILE

  // Epilogue: bias add + route to Q / K / V^T (f16).
  float bv[2];
  int   ng[2];
#pragma unroll
  for (int fn = 0; fn < 2; ++fn) {
    ng[fn] = nB + wn * 32 + fn * 16 + nl;
    bv[fn] = bias[ng[fn]];
  }
#pragma unroll
  for (int fm = 0; fm < 4; ++fm) {
#pragma unroll
    for (int fn = 0; fn < 2; ++fn) {
      int n_g = ng[fn];
#pragma unroll
      for (int g = 0; g < 8; ++g) {
        int   m_g = mB + wm * 64 + fm * 16 + g + half * 8;
        float v   = acc[fm][fn][g] + bv[fn];
        h16   hv  = (h16)v;
        if (n_g < CDIM) {
          Qb[(size_t)m_g * CDIM + n_g] = hv;
        } else if (n_g < 2 * CDIM) {
          Kb[(size_t)m_g * CDIM + (n_g - CDIM)] = hv;
        } else {
          int c = n_g - 2 * CDIM;
          int hh = c >> 7, dd = c & 127;
          int b = m_g >> 11, t = m_g & 2047;
          Vtb[((size_t)(b * HDIM + hh) * DDIM + dd) * TDIM + t] = hv;
        }
      }
    }
  }
}

// ---------------- phase 2: flash attention ----------------
// grid.x = B*H*(T/128); block = 256 (8 waves), each wave owns 16 query rows.
__global__ __launch_bounds__(256) void attn_k(const h16* __restrict__ Qb,
                                              const h16* __restrict__ Kb,
                                              const h16* __restrict__ Vtb,
                                              float* __restrict__ out) {
  __shared__ __align__(16) h16 pstrip[8][16 * 40];  // per-wave P tile (16x32, pad 40)

  const int tid  = threadIdx.x;
  const int lane = tid & 31;
  const int wave = tid >> 5;
  const int half = lane >> 4;
  const int nl   = lane & 15;
  const int bh   = blockIdx.x >> 4;   // 0..31 = b*16+h
  const int qblk = blockIdx.x & 15;
  const int b    = bh >> 4;
  const int h    = bh & 15;
  const int q0   = qblk * 128 + wave * 16;
  const size_t rowOff = (size_t)b * TDIM;
  const int    hOff   = h * DDIM;

  // Q as 4 A-fragments over d-slices of 32.
  v16h qf[4];
#pragma unroll
  for (int ds = 0; ds < 4; ++ds) {
    const h16* p = &Qb[(rowOff + q0 + nl) * CDIM + hOff + ds * 32 + half * 8];
    qf[ds] = frag_ld(p, p + 16);
  }

  v8f o[8];
#pragma unroll
  for (int df = 0; df < 8; ++df)
    o[df] = (v8f){0.f, 0.f, 0.f, 0.f, 0.f, 0.f, 0.f, 0.f};

  float rm[8], rl[8];
#pragma unroll
  for (int g = 0; g < 8; ++g) { rm[g] = -__builtin_inff(); rl[g] = 0.f; }

  // scale * log2(e): softmax computed in base-2 domain.
  const float sscale = 0.08838834764831845f * 1.4426950408889634f;

  h16* strip = &pstrip[wave][0];
  const int nkt = (q0 + 16 + 31) >> 5;  // causal: keys 0..q0+15

  for (int kt = 0; kt < nkt; ++kt) {
    const int kbase = kt * 32;

    // S = Q * K^T for 32 keys (2 N-fragments).
    v8f s[2];
    s[0] = (v8f){0.f, 0.f, 0.f, 0.f, 0.f, 0.f, 0.f, 0.f};
    s[1] = (v8f){0.f, 0.f, 0.f, 0.f, 0.f, 0.f, 0.f, 0.f};
#pragma unroll
    for (int j = 0; j < 2; ++j) {
#pragma unroll
      for (int ds = 0; ds < 4; ++ds) {
        const h16* p =
            &Kb[(rowOff + kbase + j * 16 + nl) * CDIM + hOff + ds * 32 + half * 8];
        v16h kf = frag_ld(p, p + 16);
        s[j] = __builtin_amdgcn_wmma_f32_16x16x32_f16(false, qf[ds], false, kf,
                                                      (short)0, s[j], false, false);
      }
    }

    // Scale + causal mask (only tiles crossing the diagonal need masking).
    const bool domask = (kbase + 31) > q0;
#pragma unroll
    for (int j = 0; j < 2; ++j) {
#pragma unroll
      for (int g = 0; g < 8; ++g) {
        float v = s[j][g] * sscale;
        if (domask) {
          int key  = kbase + j * 16 + nl;
          int qrow = q0 + g + half * 8;
          if (key > qrow) v = -1e30f;
        }
        s[j][g] = v;
      }
    }

    // Online softmax: row max (butterfly over the 16-lane N group).
    float alpha[8];
#pragma unroll
    for (int g = 0; g < 8; ++g) {
      float v = fmaxf(s[0][g], s[1][g]);
      v = fmaxf(v, __shfl_xor(v, 1, 32));
      v = fmaxf(v, __shfl_xor(v, 2, 32));
      v = fmaxf(v, __shfl_xor(v, 4, 32));
      v = fmaxf(v, __shfl_xor(v, 8, 32));
      float nm = fmaxf(rm[g], v);
      alpha[g] = fast_exp2(rm[g] - nm);
      rm[g]    = nm;
    }

    // P = exp2(S - m), row sums, write P (f16) to wave-private LDS strip.
#pragma unroll
    for (int g = 0; g < 8; ++g) {
      float p0 = fast_exp2(s[0][g] - rm[g]);
      float p1 = fast_exp2(s[1][g] - rm[g]);
      float rs = p0 + p1;
      rs += __shfl_xor(rs, 1, 32);
      rs += __shfl_xor(rs, 2, 32);
      rs += __shfl_xor(rs, 4, 32);
      rs += __shfl_xor(rs, 8, 32);
      rl[g] = rl[g] * alpha[g] + rs;
      int m = g + half * 8;
      strip[m * 40 + nl]      = (h16)p0;
      strip[m * 40 + 16 + nl] = (h16)p1;
    }

    // Rescale O accumulators.
#pragma unroll
    for (int df = 0; df < 8; ++df)
#pragma unroll
      for (int g = 0; g < 8; ++g) o[df][g] *= alpha[g];

    // O += P * V  (P: one 16x32 A-fragment; V^T rows are key-contiguous).
    {
      const h16* pp = &strip[nl * 40 + half * 8];
      v16h pa = frag_ld(pp, pp + 16);
#pragma unroll
      for (int df = 0; df < 8; ++df) {
        const h16* vp =
            &Vtb[((size_t)bh * DDIM + df * 16 + nl) * TDIM + kbase + half * 8];
        v16h vf = frag_ld(vp, vp + 16);
        o[df] = __builtin_amdgcn_wmma_f32_16x16x32_f16(false, pa, false, vf,
                                                       (short)0, o[df], false, false);
      }
    }
  }

  // Epilogue: normalize and write fp32 output [B,T,C] (C index = h*D + d).
  float rinv[8];
#pragma unroll
  for (int g = 0; g < 8; ++g) rinv[g] = 1.0f / rl[g];
#pragma unroll
  for (int df = 0; df < 8; ++df)
#pragma unroll
    for (int g = 0; g < 8; ++g) {
      int m = g + half * 8;
      out[(rowOff + q0 + m) * CDIM + hOff + df * 16 + nl] = o[df][g] * rinv[g];
    }
}

extern "C" void kernel_launch(void* const* d_in, const int* in_sizes, int n_in,
                              void* d_out, int out_size, void* d_ws, size_t ws_size,
                              hipStream_t stream) {
  (void)in_sizes; (void)n_in; (void)out_size; (void)ws_size;
  const float* x    = (const float*)d_in[0];
  const float* w    = (const float*)d_in[1];
  const float* bias = (const float*)d_in[2];
  float*       out  = (float*)d_out;

  const size_t nx = (size_t)BDIM * TDIM * CDIM;   // 8,388,608
  const size_t nw = (size_t)3 * CDIM * CDIM;      // 12,582,912

  h16* xb  = (h16*)d_ws;
  h16* wb  = xb + nx;
  h16* Qb  = wb + nw;
  h16* Kb  = Qb + nx;
  h16* Vtb = Kb + nx;
  // total workspace use: (nx + nw + 3*nx) * 2 bytes ~= 92.3 MB

  cvt_f32_to_f16_k<<<(int)(nx / 4 / 256), 256, 0, stream>>>(x, xb, (int)(nx / 4));
  cvt_f32_to_f16_k<<<(int)(nw / 4 / 256), 256, 0, stream>>>(w, wb, (int)(nw / 4));
  qkv_gemm_k<<<dim3(48, 32), 256, 0, stream>>>(xb, wb, bias, Qb, Kb, Vtb);
  attn_k<<<dim3(512), 256, 0, stream>>>(Qb, Kb, Vtb, out);
}
